// E_GCL_11742440587708
// MI455X (gfx1250) — matrix operand order repaired and checked by reading, not compiled, not run on hardware
//
#include <hip/hip_runtime.h>
#include <hip/hip_bf16.h>
#include <stdint.h>

// ---------------- types ----------------
typedef __bf16 bf16w;
typedef __attribute__((ext_vector_type(16))) __bf16 v16bf;
typedef __attribute__((ext_vector_type(8)))  __bf16 v8bf;
typedef __attribute__((ext_vector_type(4)))  __bf16 v4bf;
typedef __attribute__((ext_vector_type(8)))  float  v8f;

#define WAVES 8           // 8 wave32 per 256-thread block
#define EDGE_SMEM  223744 // bytes of dynamic LDS, edge kernel
#define NODE_SMEM  196608 // bytes of dynamic LDS, node kernel

__device__ __forceinline__ v8f wmma_bf16(v16bf a, v16bf b, v8f c) {
    return __builtin_amdgcn_wmma_f32_16x16x32_bf16(
        false, a, false, b, (short)0, c, false, false);
}

// A-fragment (16-bit, 16x32): lane holds row M=lane&15; 16 contiguous K
// kb..kb+15 with kb = kBase + (lane<16 ? 0 : 16).
__device__ __forceinline__ v16bf load_frag_a(const bf16w* base, int rowStride,
                                             int lane, int kBase) {
    const int m  = lane & 15;
    const int kb = kBase + ((lane < 16) ? 0 : 16);
    const bf16w* p = base + (size_t)m * rowStride + kb;
    union { v16bf v; v8bf h[2]; } u;
    u.h[0] = *(const v8bf*)(p);
    u.h[1] = *(const v8bf*)(p + 8);
    return u.v;
}

// B-fragment (16-bit, 32x16) from K-major weights WT[n][k] (in LDS):
// lane holds col N = nBase + (lane&15), 16 contiguous K values.
__device__ __forceinline__ v16bf load_frag_b(const bf16w* wt, int kStride,
                                             int lane, int nBase, int kBase) {
    const int n = nBase + (lane & 15);
    const int k = kBase + ((lane < 16) ? 0 : 16);
    return *(const v16bf*)(wt + (size_t)n * kStride + k);
}

__device__ __forceinline__ float silu(float x) {
    return x * (1.0f / (1.0f + __expf(-x)));
}

// ---- CDNA5 async copy global->LDS (ASYNCcnt path) ----
__device__ __forceinline__ void async_copy_b128(uint32_t lds_off, const void* gptr) {
    asm volatile("global_load_async_to_lds_b128 %0, %1, off"
                 :: "v"(lds_off), "v"((uint64_t)(uintptr_t)gptr)
                 : "memory");
}
__device__ __forceinline__ void wait_async0() {
    asm volatile("s_wait_asynccnt 0x0" ::: "memory");
}
// Wave-local LDS producer->consumer sync (DS ops are in-order per wave).
__device__ __forceinline__ void wave_lds_sync() {
    asm volatile("s_wait_dscnt 0x0" ::: "memory");
}

// ---------------- prep: transpose + convert weights to bf16 in ws ----------------
__global__ void prep_weights(const float* __restrict__ eW1, const float* __restrict__ eW2,
                             const float* __restrict__ cW1, const float* __restrict__ nW1,
                             const float* __restrict__ nW2,
                             bf16w* __restrict__ WT1, bf16w* __restrict__ WT2,
                             bf16w* __restrict__ cW1T, bf16w* __restrict__ nW1T,
                             bf16w* __restrict__ nW2T) {
    int i = blockIdx.x * blockDim.x + threadIdx.x;
    if (i < 128 * 288) {                       // WT1 [128][288] from eW1 [4][268][32]
        int n = i / 288, k = i % 288;
        int head = n >> 5, nl = n & 31;
        float v = (k < 268) ? eW1[(size_t)head * 268 * 32 + (size_t)k * 32 + nl] : 0.0f;
        WT1[i] = (bf16w)v;
        return;
    }
    i -= 128 * 288;
    if (i < 128 * 32) {                        // WT2 [128][32] from eW2 [4][32][32]
        int n = i / 32, k = i % 32;
        int head = n >> 5, nl = n & 31;
        WT2[i] = (bf16w)eW2[(size_t)head * 1024 + (size_t)k * 32 + nl];
        return;
    }
    i -= 128 * 32;
    if (i < 128 * 128) {                       // cW1T
        int n = i / 128, k = i % 128;
        cW1T[i] = (bf16w)cW1[(size_t)k * 128 + n];
        return;
    }
    i -= 128 * 128;
    if (i < 128 * 256) {                       // nW1T
        int n = i / 256, k = i % 256;
        nW1T[i] = (bf16w)nW1[(size_t)k * 128 + n];
        return;
    }
    i -= 128 * 256;
    if (i < 128 * 128) {                       // nW2T
        int n = i / 128, k = i % 128;
        nW2T[i] = (bf16w)nW2[(size_t)k * 128 + n];
        return;
    }
}

// ---------------- init: agg = 0, coord_out = coord ----------------
__global__ void init_buffers(float* __restrict__ agg, float* __restrict__ coord_out,
                             const float* __restrict__ coord, long n_agg, long n_coord) {
    long i = (long)blockIdx.x * blockDim.x + threadIdx.x;
    if (i < n_agg)   agg[i] = 0.0f;
    if (i < n_coord) coord_out[i] = coord[i];
}

// Pipelined 16xN GEMM over LDS operands: K-chunk outer (A reused across all
// 8 N tiles), B fragments rotated one ahead of the consuming WMMA so the
// per-WMMA s_wait_dscnt has slack instead of draining to zero.
__device__ __forceinline__ void gemm_16xN(v8f acc[8], const bf16w* Abase, int aStride,
                                          const bf16w* Wt, int kStride, int lane,
                                          int nChunks) {
    v16bf a = load_frag_a(Abase, aStride, lane, 0);
    for (int kc = 0; kc < nChunks; ++kc) {
        v16bf b = load_frag_b(Wt, kStride, lane, 0, kc * 32);
        const v16bf a_next = (kc + 1 < nChunks)
            ? load_frag_a(Abase, aStride, lane, (kc + 1) * 32) : a;
        #pragma unroll
        for (int t = 0; t < 8; ++t) {
            const v16bf b_next = (t < 7)
                ? load_frag_b(Wt, kStride, lane, (t + 1) * 16, kc * 32) : b;
            acc[t] = wmma_bf16(a, b, acc[t]);
            b = b_next;
        }
        a = a_next;
    }
}

// ---------------- edge kernel ----------------
// Dynamic LDS layout (bytes):
//   0      : weights WT1|WT2|cW1T  (57344 bf16 = 114688 B, LDS-resident for block)
//   114688 : feats  [8 waves][16*288 bf16]  (9216 B each)
//   188416 : astage [8 waves][16*128 bf16]  (4096 B each)
//   221184 : s_row  [8][16] int
//   221696 : s_col  [8][16] int
//   222208 : s_cd   [8][16*3] float
__global__ __launch_bounds__(256) void edge_kernel(
    const float* __restrict__ h, const float* __restrict__ coord,
    const int* __restrict__ eidx,
    const float* __restrict__ eb1, const float* __restrict__ eb2,
    const float* __restrict__ ln_g, const float* __restrict__ ln_b,
    const float* __restrict__ cb1, const float* __restrict__ cW2,
    const bf16w* __restrict__ Wg,   // WT1|WT2|cW1T contiguous, 57344 bf16
    float* __restrict__ agg, float* __restrict__ coord_out, int E, int nb) {

    extern __shared__ char smem[];
    bf16w* sWT1  = (bf16w*)smem;             // [128][288]
    bf16w* sWT2  = sWT1 + 36864;             // [128][32]
    bf16w* scW1  = sWT1 + 40960;             // [128][128]

    const int wave = threadIdx.x >> 5;
    const int lane = threadIdx.x & 31;
    const int nlane = lane & 15;

    bf16w* sfeats = (bf16w*)(smem + 114688) + wave * 4608;
    bf16w* sast   = (bf16w*)(smem + 188416) + wave * 2048;
    int*   srow   = (int*)  (smem + 221184) + wave * 16;
    int*   scol   = (int*)  (smem + 221696) + wave * 16;
    float* scd    = (float*)(smem + 222208) + wave * 48;

    // ---- one-time: async-stage all edge weights into LDS (114688 B = 7168 x b128) ----
    for (int i = threadIdx.x; i < 7168; i += 256) {
        async_copy_b128((uint32_t)(uintptr_t)(smem + i * 16),
                        (const char*)Wg + i * 16);
    }
    wait_async0();
    __syncthreads();

    for (int base = blockIdx.x; base < nb; base += gridDim.x) {
        const long e0 = ((long)base * WAVES + wave) * 16;

        // ---- stage 0: per-edge geometry (lanes 0..15) ----
        if (lane < 16) {
            const long e = e0 + lane;
            const bool valid = (e < (long)E);
            int r = 0, c = 0;
            if (valid) { r = eidx[e]; c = eidx[(long)E + e]; }
            const float cix = coord[(long)r * 3 + 0], ciy = coord[(long)r * 3 + 1], ciz = coord[(long)r * 3 + 2];
            const float ckx = coord[(long)c * 3 + 0], cky = coord[(long)c * 3 + 1], ckz = coord[(long)c * 3 + 2];
            const float dx = cix - ckx, dy = ciy - cky, dz = ciz - ckz;
            const float radial = dx * dx + dy * dy + dz * dz;
            const float dist = sqrtf(radial);
            const float dotv = cix * ckx + ciy * cky + ciz * ckz;
            const float eps = 1e-8f, thr = 1e-6f;
            const float ia = 1.0f / (dist + eps);
            float ax = dx * ia, ay = dy * ia, az = dz * ia;
            float cpx = ciy * ckz - ciz * cky;
            float cpy = ciz * ckx - cix * ckz;
            float cpz = cix * cky - ciy * ckx;
            const float ib = 1.0f / (sqrtf(cpx * cpx + cpy * cpy + cpz * cpz) + eps);
            float bx = cpx * ib, by = cpy * ib, bz = cpz * ib;
            float cx = ay * bz - az * by;
            float cy = az * bx - ax * bz;
            float cz = ax * by - ay * bx;
            const float na = sqrtf(ax * ax + ay * ay + az * az);
            const float nbv = sqrtf(bx * bx + by * by + bz * bz);
            const float nc = sqrtf(cx * cx + cy * cy + cz * cz);
            if ((na < thr) || (nbv < thr) || (nc < thr)) {
                ax = 1.f; bx = 0.f; cx = 0.f;
                ay = 0.f; by = 1.f; cy = 0.f;
                az = 0.f; bz = 0.f; cz = 1.f;
            }
            bf16w* fp = sfeats + lane * 288;
            fp[256] = (bf16w)radial; fp[257] = (bf16w)dist; fp[258] = (bf16w)dotv;
            fp[259] = (bf16w)ax; fp[260] = (bf16w)bx; fp[261] = (bf16w)cx;
            fp[262] = (bf16w)ay; fp[263] = (bf16w)by; fp[264] = (bf16w)cy;
            fp[265] = (bf16w)az; fp[266] = (bf16w)bz; fp[267] = (bf16w)cz;
            #pragma unroll
            for (int j = 268; j < 288; ++j) fp[j] = (bf16w)0.0f;
            srow[lane] = valid ? r : -1;
            scol[lane] = c;
            scd[lane * 3 + 0] = dx; scd[lane * 3 + 1] = dy; scd[lane * 3 + 2] = dz;
        }
        wave_lds_sync();

        // ---- stage 1: gather h[row], h[col] -> feats ----
        for (int e = 0; e < 16; ++e) {
            const int r0 = srow[e];
            const int rr = (r0 < 0) ? 0 : r0;
            const int cc = scol[e];
            const float4 hv = *(const float4*)(h + (long)rr * 128 + lane * 4);
            const float4 cv = *(const float4*)(h + (long)cc * 128 + lane * 4);
            v4bf hb; hb[0] = (bf16w)hv.x; hb[1] = (bf16w)hv.y; hb[2] = (bf16w)hv.z; hb[3] = (bf16w)hv.w;
            v4bf cb; cb[0] = (bf16w)cv.x; cb[1] = (bf16w)cv.y; cb[2] = (bf16w)cv.z; cb[3] = (bf16w)cv.w;
            *(v4bf*)(sfeats + e * 288 + lane * 4)       = hb;
            *(v4bf*)(sfeats + e * 288 + 128 + lane * 4) = cb;
        }
        wave_lds_sync();

        // ---- stage 2: layer-1 [16,288]x[288,128] + SiLU -> sast (bf16) ----
        {
            v8f acc[8] = {};
            gemm_16xN(acc, sfeats, 288, sWT1, 288, lane, 9);
            #pragma unroll
            for (int t = 0; t < 8; ++t) {
                const int n = t * 16 + nlane;
                const float bias = eb1[n];
                #pragma unroll
                for (int i = 0; i < 8; ++i) {
                    const int m = i + ((lane < 16) ? 0 : 8);
                    sast[m * 128 + n] = (bf16w)silu(acc[t][i] + bias);
                }
            }
        }
        wave_lds_sync();

        // ---- stage 3: per-head layer-2 [16,32]x[32,32] + bias -> regs ----
        v8f ef[8];
        #pragma unroll
        for (int head = 0; head < 4; ++head) {
            const v16bf a = load_frag_a(sast + head * 32, 128, lane, 0);
            #pragma unroll
            for (int tt = 0; tt < 2; ++tt) {
                const int t = head * 2 + tt;
                const v16bf b = load_frag_b(sWT2, 32, lane, t * 16, 0);
                v8f acc = {};
                acc = wmma_bf16(a, b, acc);
                const float bias = eb2[t * 16 + nlane];
                #pragma unroll
                for (int i = 0; i < 8; ++i) ef[t][i] = acc[i] + bias;
            }
        }

        // ---- stage 4: LayerNorm across 128 (row M in one 16-lane half) ----
        float mu[8], rs[8];
        #pragma unroll
        for (int i = 0; i < 8; ++i) {
            float s = 0.f, q = 0.f;
            #pragma unroll
            for (int t = 0; t < 8; ++t) { const float v = ef[t][i]; s += v; q += v * v; }
            #pragma unroll
            for (int msk = 1; msk < 16; msk <<= 1) {
                s += __shfl_xor(s, msk);
                q += __shfl_xor(q, msk);
            }
            const float mean = s * (1.0f / 128.0f);
            mu[i] = mean;
            rs[i] = rsqrtf(q * (1.0f / 128.0f) - mean * mean + 1e-5f);
        }
        #pragma unroll
        for (int t = 0; t < 8; ++t) {
            const int n = t * 16 + nlane;
            const float g = ln_g[n], bb = ln_b[n];
            #pragma unroll
            for (int i = 0; i < 8; ++i) {
                const float v = (ef[t][i] - mu[i]) * rs[i] * g + bb;
                const int m = i + ((lane < 16) ? 0 : 8);
                sast[m * 128 + n] = (bf16w)v;
                const int r = srow[m];
                if (r >= 0) atomicAdd(&agg[(long)r * 128 + n], v);
            }
        }
        wave_lds_sync();

        // ---- stage 5: coord MLP: w = silu(ef @ cW1 + cb1) . cW2 ----
        float wacc[8] = {0.f, 0.f, 0.f, 0.f, 0.f, 0.f, 0.f, 0.f};
        {
            v8f acc[8] = {};
            gemm_16xN(acc, sast, 128, scW1, 128, lane, 4);
            #pragma unroll
            for (int t = 0; t < 8; ++t) {
                const int n = t * 16 + nlane;
                const float bias = cb1[n];
                const float w2 = cW2[n];
                #pragma unroll
                for (int i = 0; i < 8; ++i) wacc[i] += silu(acc[t][i] + bias) * w2;
            }
        }
        #pragma unroll
        for (int i = 0; i < 8; ++i) {
            #pragma unroll
            for (int msk = 1; msk < 16; msk <<= 1) wacc[i] += __shfl_xor(wacc[i], msk);
        }
        if (nlane == 0) {
            const int halfbase = (lane < 16) ? 0 : 8;
            #pragma unroll
            for (int i = 0; i < 8; ++i) {
                const int m = halfbase + i;
                const int r = srow[m];
                if (r >= 0) {
                    const float w = wacc[i];
                    atomicAdd(&coord_out[(long)r * 3 + 0], scd[m * 3 + 0] * w);
                    atomicAdd(&coord_out[(long)r * 3 + 1], scd[m * 3 + 1] * w);
                    atomicAdd(&coord_out[(long)r * 3 + 2], scd[m * 3 + 2] * w);
                }
            }
        }
        wave_lds_sync();   // protect srow/scd/sast before next iteration overwrites
    }
}

// ---------------- node kernel ----------------
// Dynamic LDS layout (bytes):
//   0      : weights nW1T|nW2T (49152 bf16 = 98304 B)
//   98304  : s_in  [8 waves][16*256 bf16]  (8192 B each)
//   163840 : s_mid [8 waves][16*128 bf16]  (4096 B each)
__global__ __launch_bounds__(256) void node_kernel(
    const float* __restrict__ h, const float* __restrict__ agg,
    const float* __restrict__ nb1, const float* __restrict__ nb2,
    const bf16w* __restrict__ Wg,   // nW1T|nW2T contiguous, 49152 bf16
    float* __restrict__ h_out, int N, int nb) {

    extern __shared__ char smem[];
    bf16w* sNW1 = (bf16w*)smem;        // [128][256]
    bf16w* sNW2 = sNW1 + 32768;        // [128][128]

    const int wave = threadIdx.x >> 5;
    const int lane = threadIdx.x & 31;
    const int nlane = lane & 15;

    bf16w* s_in  = (bf16w*)(smem + 98304)  + wave * 4096;
    bf16w* s_mid = (bf16w*)(smem + 163840) + wave * 2048;

    // one-time async weight staging: 98304 B = 6144 x b128
    for (int i = threadIdx.x; i < 6144; i += 256) {
        async_copy_b128((uint32_t)(uintptr_t)(smem + i * 16),
                        (const char*)Wg + i * 16);
    }
    wait_async0();
    __syncthreads();

    for (int base = blockIdx.x; base < nb; base += gridDim.x) {
        const long n0 = ((long)base * WAVES + wave) * 16;

        for (int e = 0; e < 16; ++e) {
            const long node = n0 + e;
            const long nn = (node < (long)N) ? node : 0;
            const float4 hv = *(const float4*)(h + nn * 128 + lane * 4);
            const float4 av = *(const float4*)(agg + nn * 128 + lane * 4);
            v4bf hb; hb[0] = (bf16w)hv.x; hb[1] = (bf16w)hv.y; hb[2] = (bf16w)hv.z; hb[3] = (bf16w)hv.w;
            v4bf ab; ab[0] = (bf16w)av.x; ab[1] = (bf16w)av.y; ab[2] = (bf16w)av.z; ab[3] = (bf16w)av.w;
            *(v4bf*)(s_in + e * 256 + lane * 4)       = hb;
            *(v4bf*)(s_in + e * 256 + 128 + lane * 4) = ab;
        }
        wave_lds_sync();

        // GEMM1: [16,256] x [256,128] + SiLU -> s_mid
        {
            v8f acc[8] = {};
            gemm_16xN(acc, s_in, 256, sNW1, 256, lane, 8);
            #pragma unroll
            for (int t = 0; t < 8; ++t) {
                const int n = t * 16 + nlane;
                const float bias = nb1[n];
                #pragma unroll
                for (int i = 0; i < 8; ++i) {
                    const int m = i + ((lane < 16) ? 0 : 8);
                    s_mid[m * 128 + n] = (bf16w)silu(acc[t][i] + bias);
                }
            }
        }
        wave_lds_sync();

        // GEMM2: [16,128] x [128,128] + bias + residual -> h_out
        {
            v8f acc[8] = {};
            gemm_16xN(acc, s_mid, 128, sNW2, 128, lane, 4);
            #pragma unroll
            for (int t = 0; t < 8; ++t) {
                const int n = t * 16 + nlane;
                const float bias = nb2[n];
                #pragma unroll
                for (int i = 0; i < 8; ++i) {
                    const int m = i + ((lane < 16) ? 0 : 8);
                    const long node = n0 + m;
                    if (node < (long)N)
                        h_out[node * 128 + n] = h[node * 128 + n] + acc[t][i] + bias;
                }
            }
        }
        wave_lds_sync();
    }
}

// ---------------- host launch ----------------
extern "C" void kernel_launch(void* const* d_in, const int* in_sizes, int n_in,
                              void* d_out, int out_size, void* d_ws, size_t ws_size,
                              hipStream_t stream) {
    const float* h      = (const float*)d_in[0];
    const float* coord  = (const float*)d_in[1];
    const int*   eidx   = (const int*)d_in[2];
    const float* eW1    = (const float*)d_in[3];
    const float* eb1    = (const float*)d_in[4];
    const float* eW2    = (const float*)d_in[5];
    const float* eb2    = (const float*)d_in[6];
    const float* ln_g   = (const float*)d_in[7];
    const float* ln_b   = (const float*)d_in[8];
    const float* nW1    = (const float*)d_in[9];
    const float* nb1    = (const float*)d_in[10];
    const float* nW2    = (const float*)d_in[11];
    const float* nb2    = (const float*)d_in[12];
    const float* cW1    = (const float*)d_in[13];
    const float* cb1    = (const float*)d_in[14];
    const float* cW2    = (const float*)d_in[15];

    const int N = in_sizes[0] / 128;
    const int E = in_sizes[2] / 2;

    float* h_out     = (float*)d_out;
    float* coord_out = h_out + (size_t)N * 128;

    // workspace: agg | WT1 | WT2 | cW1T | nW1T | nW2T  (edge & node weight
    // groups each contiguous so kernels can async-copy one region to LDS)
    uint8_t* ws = (uint8_t*)d_ws;
    float* agg = (float*)ws;
    size_t off = (size_t)N * 128 * sizeof(float);
    bf16w* WT1  = (bf16w*)(ws + off); off += (size_t)128 * 288 * 2;
    bf16w* WT2  = (bf16w*)(ws + off); off += (size_t)128 * 32 * 2;
    bf16w* cW1T = (bf16w*)(ws + off); off += (size_t)128 * 128 * 2;
    bf16w* nW1T = (bf16w*)(ws + off); off += (size_t)128 * 256 * 2;
    bf16w* nW2T = (bf16w*)(ws + off); off += (size_t)128 * 128 * 2;

    (void)hipFuncSetAttribute((const void*)edge_kernel,
                              hipFuncAttributeMaxDynamicSharedMemorySize, EDGE_SMEM);
    (void)hipFuncSetAttribute((const void*)node_kernel,
                              hipFuncAttributeMaxDynamicSharedMemorySize, NODE_SMEM);

    const int prep_total = 128 * 288 + 128 * 32 + 128 * 128 + 128 * 256 + 128 * 128;
    prep_weights<<<(prep_total + 255) / 256, 256, 0, stream>>>(
        eW1, eW2, cW1, nW1, nW2, WT1, WT2, cW1T, nW1T, nW2T);

    const long n_agg = (long)N * 128;
    init_buffers<<<(int)((n_agg + 255) / 256), 256, 0, stream>>>(
        agg, coord_out, coord, n_agg, (long)N * 3);

    const int etiles = (E + 15) / 16;
    const int nb_e = (etiles + WAVES - 1) / WAVES;
    const int egrid = (nb_e < 1024) ? nb_e : 1024;
    edge_kernel<<<egrid, 256, EDGE_SMEM, stream>>>(
        h, coord, eidx, eb1, eb2, ln_g, ln_b, cb1, cW2,
        WT1, agg, coord_out, E, nb_e);

    const int ntiles = (N + 15) / 16;
    const int nb_n = (ntiles + WAVES - 1) / WAVES;
    const int ngrid = (nb_n < 512) ? nb_n : 512;
    node_kernel<<<ngrid, 256, NODE_SMEM, stream>>>(
        h, agg, nb1, nb2, nW1T, h_out, N, nb_n);
}
